// YOLOv2_26456998543455
// MI455X (gfx1250) — matrix operand order repaired and checked by reading, not compiled, not run on hardware
//
#include <hip/hip_runtime.h>

// ---------------------------------------------------------------- types
typedef __bf16 bf16_t;
typedef __attribute__((ext_vector_type(16))) __bf16 v16bf;
typedef __attribute__((ext_vector_type(8)))  __bf16 v8bf;
typedef __attribute__((ext_vector_type(8)))  float  v8f;

// async copy builtin wants pointers to 16-byte int vectors in AS1/AS3
typedef int v4i_b128 __attribute__((vector_size(16)));
typedef __attribute__((address_space(1))) v4i_b128* as1_v4ip;
typedef __attribute__((address_space(3))) v4i_b128* as3_v4ip;

#ifndef __has_builtin
#define __has_builtin(x) 0
#endif
#if __has_builtin(__builtin_amdgcn_global_load_async_to_lds_b128) && \
    __has_builtin(__builtin_amdgcn_s_wait_asynccnt)
#define USE_ASYNC_LDS 1
#endif

// ---------------------------------------------------------------- problem dims
// STRIDE=32, INPUT=1280, FMP=40, HW=1600, KA=5, NC=20, pred ch = 125
#define HW40    1600
#define NPRED   125
#define NSCORES 160000
#define TOPK    100

// ---------------------------------------------------------------- workspace layout
__host__ __device__ constexpr size_t AL(size_t x) { return (x + 255) & ~size_t(255); }
constexpr size_t OFF_WB1A   = 0;
constexpr size_t OFF_WB1B   = AL(OFF_WB1A   + 1024ull * 9216 * 2);
constexpr size_t OFF_WB2    = AL(OFF_WB1B   + 1024ull * 9216 * 2);
constexpr size_t OFF_ACT0   = AL(OFF_WB2    + 1024ull * 11520 * 2);
constexpr size_t OFF_ACT1   = AL(OFF_ACT0   + 1600ull * 1024 * 2);
constexpr size_t OFF_ACT3   = AL(OFF_ACT1   + 1600ull * 1024 * 2);   // concat [1600][1280]
constexpr size_t OFF_ACT4   = AL(OFF_ACT3   + 1600ull * 1280 * 2);
constexpr size_t OFF_PRED   = AL(OFF_ACT4   + 1600ull * 1024 * 2);
constexpr size_t OFF_SCORES = AL(OFF_PRED   + 1600ull * 125  * 4);
constexpr size_t OFF_ZROW   = AL(OFF_SCORES + 160000ull * 4);        // 1280 bf16 zeros

// ---------------------------------------------------------------- small helpers
__global__ void zero_bf16(bf16_t* __restrict__ p, int n)
{
    int i = blockIdx.x * blockDim.x + threadIdx.x;
    if (i < n) p[i] = (bf16_t)0.f;
}

// w OIHW fp32 -> wB[o][tap*Cin + c] bf16, BN scale folded in.
__global__ void prep_weights(const float* __restrict__ w, const float* __restrict__ scale,
                             bf16_t* __restrict__ outW, int Cin, int total)
{
    int i = blockIdx.x * blockDim.x + threadIdx.x;
    if (i >= total) return;
    int o   = i / (Cin * 9);
    int r   = i % (Cin * 9);
    int c   = r / 9;
    int tap = r % 9;
    outW[(size_t)o * (9 * Cin) + (size_t)tap * Cin + c] = (bf16_t)(w[i] * scale[o]);
}

// c5 NCHW fp32 -> act0 [HW][C] bf16
__global__ void nchw_to_hwc_bf16(const float* __restrict__ in, bf16_t* __restrict__ out,
                                 int C, int HW, int total)
{
    int i = blockIdx.x * blockDim.x + threadIdx.x;
    if (i >= total) return;
    int ch = i / HW, p = i % HW;
    out[(size_t)p * C + ch] = (bf16_t)in[i];
}

// ---------------------------------------------------------------- reorg path
// 1x1 conv 512->64 on 80x80 + BN + lrelu, then reorg(2) -> cols [0,256) of act3 [1600][1280]
__global__ void reorg_concat(const float* __restrict__ c4, const float* __restrict__ wr,
                             const float* __restrict__ sr, const float* __restrict__ br,
                             bf16_t* __restrict__ act3)
{
    int i = blockIdx.x * blockDim.x + threadIdx.x;
    if (i >= 1600 * 256) return;
    int ch = i & 255, p = i >> 8;
    int q  = ch >> 6, c = ch & 63;
    int si = q >> 1,  sj = q & 1;
    int yy = p / 40,  xx = p % 40;
    int Y = 2 * yy + si, X = 2 * xx + sj;
    const float* col  = c4 + (size_t)Y * 80 + X;
    const float* wrow = wr + (size_t)c * 512;
    float s = 0.f;
    for (int k = 0; k < 512; ++k) s += col[(size_t)k * 6400] * wrow[k];
    s = s * sr[c] + br[c];
    s = s > 0.f ? s : 0.1f * s;
    act3[(size_t)p * 1280 + ch] = (bf16_t)s;
}

// ---------------------------------------------------------------- WMMA implicit-GEMM 3x3 conv
// act [1600][CIN] bf16, weights [1024][9*CIN] bf16 (tap-major K => contiguous), bias fp32.
// Double-buffered LDS B tiles filled by async-to-LDS; out = leaky_relu(gemm + bias) as bf16.
template <int CIN>
__global__ __launch_bounds__(256)
void conv3x3_wmma(const bf16_t* __restrict__ actIn, const bf16_t* __restrict__ wB,
                  const float* __restrict__ bias, const bf16_t* __restrict__ zrow,
                  bf16_t* __restrict__ out, int outStride, int outColOff)
{
    constexpr int K9     = 9 * CIN;
    constexpr int KPT    = CIN / 32;       // K-steps per tap
    constexpr int KSTEPS = 9 * KPT;

    __shared__ alignas(64) bf16_t ldsB[2][128 * 32];   // two 8 KB B tiles (128 N x 32 K)

    const int tid  = threadIdx.x;
    const int lane = tid & 31;
    const int wave = tid >> 5;
    const int wm   = wave & 3;      // 4 waves along M
    const int wn   = wave >> 2;     // 2 waves along N
    const int m0   = blockIdx.x * 64;
    const int n0   = blockIdx.y * 128;
    const int h    = lane >> 4;     // half-wave selector
    const int rl   = lane & 15;

    const int row = m0 + wm * 16 + rl;   // output pixel handled by this lane (A row)
    const int py  = row / 40, px = row % 40;

    // ---- per-thread B staging coords: 512 16B chunks, thread t owns chunks t and t+256
    const int n1   = tid >> 2;          // N row of first chunk (0..63)
    const int part = tid & 3;           // 16B part within the 64B K-row
    const bf16_t* srcBase1 = wB + (size_t)(n0 + n1) * K9 + part * 8;
    const bf16_t* srcBase2 = srcBase1 + (size_t)64 * K9;          // N row + 64
    const int dstOff1 = n1 * 32 + part * 8;
    const int dstOff2 = dstOff1 + 64 * 32;

    auto stage = [&](int s, int buf) {
        const bf16_t* s1 = srcBase1 + 32 * s;   // K layout is contiguous across taps
        const bf16_t* s2 = srcBase2 + 32 * s;
        bf16_t* t1 = &ldsB[buf][dstOff1];
        bf16_t* t2 = &ldsB[buf][dstOff2];
#ifdef USE_ASYNC_LDS
        __builtin_amdgcn_global_load_async_to_lds_b128(
            (as1_v4ip)(void*)s1, (as3_v4ip)(void*)t1, 0, 0);
        __builtin_amdgcn_global_load_async_to_lds_b128(
            (as1_v4ip)(void*)s2, (as3_v4ip)(void*)t2, 0, 0);
#else
        *(v8bf*)t1 = *(const v8bf*)s1;
        *(v8bf*)t2 = *(const v8bf*)s2;
#endif
    };

    v8f acc[4];
    for (int j = 0; j < 4; ++j)
        for (int e = 0; e < 8; ++e) acc[j][e] = 0.f;

    stage(0, 0);

    #pragma unroll 1
    for (int s = 0; s < KSTEPS; ++s) {
        const int buf = s & 1;
        if (s + 1 < KSTEPS) stage(s + 1, buf ^ 1);   // prefetch next B tile into other buffer

        // ---- A fragment (16x32 bf16): halo lanes read a zero row (no per-element masking)
        const int tap = s / KPT;
        const int kc  = (s - tap * KPT) * 32;
        const int dy  = tap / 3 - 1, dx = tap % 3 - 1;
        const int ny  = py + dy, nx = px + dx;
        const bool valid = ((unsigned)ny < 40u) & ((unsigned)nx < 40u);
        const bf16_t* arow = valid ? (actIn + (size_t)(ny * 40 + nx) * CIN) : zrow;

        v16bf afrag;
        {
            v8bf lo = *(const v8bf*)(arow + kc + 16 * h);
            v8bf hi = *(const v8bf*)(arow + kc + 16 * h + 8);
            #pragma unroll
            for (int e = 0; e < 8; ++e) { afrag[e] = lo[e]; afrag[8 + e] = hi[e]; }
        }

#ifdef USE_ASYNC_LDS
        if (s + 1 < KSTEPS) __builtin_amdgcn_s_wait_asynccnt(2);  // current tile landed
        else                __builtin_amdgcn_s_wait_asynccnt(0);
#endif
        __syncthreads();

        // ---- 4 WMMAs: N subtiles share the A fragment
        #pragma unroll
        for (int j = 0; j < 4; ++j) {
            const int nl = wn * 64 + j * 16 + rl;   // B column = lane%16 within subtile
            v16bf bfrag = *(const v16bf*)(&ldsB[buf][nl * 32 + 16 * h]);
            acc[j] = __builtin_amdgcn_wmma_f32_16x16x32_bf16(
                false, afrag, false, bfrag, (short)0, acc[j], false, false);
        }
        __syncthreads();   // reads done before this buffer is refilled at s+2
    }

    // ---- epilogue: + shift, leaky relu, store bf16.  C/D layout: VGPR r -> M=r+8h, N=lane%16
    for (int j = 0; j < 4; ++j) {
        int col = n0 + wn * 64 + j * 16 + rl;
        float b = bias[col];
        #pragma unroll
        for (int r = 0; r < 8; ++r) {
            int m = m0 + wm * 16 + 8 * h + r;
            float v = acc[j][r] + b;
            v = v > 0.f ? v : 0.1f * v;
            out[(size_t)m * outStride + outColOff + col] = (bf16_t)v;
        }
    }
}

// ---------------------------------------------------------------- pred 1x1 conv (1024 -> 125)
__global__ void pred_conv(const bf16_t* __restrict__ act4, const float* __restrict__ wp,
                          const float* __restrict__ bp, float* __restrict__ pred)
{
    int i = blockIdx.x * blockDim.x + threadIdx.x;
    if (i >= HW40 * NPRED) return;
    int o = i % NPRED, p = i / NPRED;
    const bf16_t* a = act4 + (size_t)p * 1024;
    const float*  w = wp   + (size_t)o * 1024;
    float s = bp[o];
    for (int c = 0; c < 1024; ++c) s += (float)a[c] * w[c];
    pred[(size_t)p * NPRED + o] = s;
}

// ---------------------------------------------------------------- scores: sigmoid(conf)*softmax(cls)
__global__ void score_kernel(const float* __restrict__ pred, float* __restrict__ scores)
{
    int A = blockIdx.x * blockDim.x + threadIdx.x;   // anchor index = pixel*5 + a
    if (A >= 8000) return;
    int pix = A / 5, a = A % 5;
    const float* rowp = pred + (size_t)pix * NPRED;
    float conf = 1.f / (1.f + expf(-rowp[a]));
    float l[20], mx = -1e30f;
    for (int c = 0; c < 20; ++c) { l[c] = rowp[5 + a * 20 + c]; mx = fmaxf(mx, l[c]); }
    float sum = 0.f;
    for (int c = 0; c < 20; ++c) { l[c] = expf(l[c] - mx); sum += l[c]; }
    float inv = conf / sum;
    for (int c = 0; c < 20; ++c) scores[(size_t)A * 20 + c] = l[c] * inv;
}

// ---------------------------------------------------------------- top-100 + decode + NMS + output
__global__ __launch_bounds__(1024)
void post_kernel(const float* __restrict__ pred, float* __restrict__ scores,
                 float* __restrict__ outp)
{
    const int tid = threadIdx.x;
    __shared__ float rs[1024];
    __shared__ int   ri[1024];
    __shared__ float tsc[TOPK];
    __shared__ int   tix[TOPK];
    __shared__ float bx[TOPK][4];
    __shared__ int   lab[TOPK];
    __shared__ int   keep[TOPK];

    // ---- iterative top-100 (destroys ws scores; rebuilt every call by score_kernel)
    for (int k = 0; k < TOPK; ++k) {
        float bv = -1e30f; int bi = 0x7fffffff;
        for (int i = tid; i < NSCORES; i += 1024) {
            float v = scores[i];
            if (v > bv) { bv = v; bi = i; }
        }
        rs[tid] = bv; ri[tid] = bi;
        __syncthreads();
        for (int s = 512; s > 0; s >>= 1) {
            if (tid < s) {
                bool take = (rs[tid + s] > rs[tid]) ||
                            (rs[tid + s] == rs[tid] && ri[tid + s] < ri[tid]);
                if (take) { rs[tid] = rs[tid + s]; ri[tid] = ri[tid + s]; }
            }
            __syncthreads();
        }
        if (tid == 0) { tsc[k] = rs[0]; tix[k] = ri[0]; scores[ri[0]] = -1e30f; }
        __syncthreads();
    }

    // ---- decode boxes
    if (tid < TOPK) {
        const float AW[5] = {1.19f, 2.79f, 4.53f, 8.06f, 10.32f};
        const float AH[5] = {1.98f, 4.59f, 8.92f, 5.29f, 10.65f};
        int id   = tix[tid];
        int aidx = id / 20;
        lab[tid] = id % 20;
        int pix = aidx / 5, a = aidx % 5;
        float gx = (float)(pix % 40), gy = (float)(pix / 40);
        const float* rg = pred + (size_t)pix * NPRED + 105 + a * 4;
        float cx = (1.f / (1.f + expf(-rg[0])) + gx) * 32.f;
        float cy = (1.f / (1.f + expf(-rg[1])) + gy) * 32.f;
        float w  = expf(rg[2]) * AW[a] * 32.f;
        float hh = expf(rg[3]) * AH[a] * 32.f;
        bx[tid][0] = cx - 0.5f * w;  bx[tid][1] = cy - 0.5f * hh;
        bx[tid][2] = cx + 0.5f * w;  bx[tid][3] = cy + 0.5f * hh;
        keep[tid] = tsc[tid] > 0.001f ? 1 : 0;
    }
    __syncthreads();

    // ---- class-aware greedy NMS (serial, 100x100)
    if (tid == 0) {
        for (int i = 0; i < TOPK; ++i) {
            if (!keep[i]) continue;
            float ax1 = bx[i][0], ay1 = bx[i][1], ax2 = bx[i][2], ay2 = bx[i][3];
            float areaA = (ax2 - ax1) * (ay2 - ay1);
            for (int j = i + 1; j < TOPK; ++j) {
                if (!keep[j] || lab[j] != lab[i]) continue;
                float xx1 = fmaxf(ax1, bx[j][0]), yy1 = fmaxf(ay1, bx[j][1]);
                float xx2 = fminf(ax2, bx[j][2]), yy2 = fminf(ay2, bx[j][3]);
                float inter = fmaxf(1e-10f, xx2 - xx1) * fmaxf(1e-10f, yy2 - yy1);
                float areaB = (bx[j][2] - bx[j][0]) * (bx[j][3] - bx[j][1]);
                float iou = inter / (areaA + areaB - inter);
                if (iou > 0.6f) keep[j] = 0;
            }
        }
    }
    __syncthreads();

    // ---- outputs: [bboxes 400 | scores 100 | labels 100 | keep 100]
    if (tid < TOPK) {
        int kp = keep[tid];
        for (int r = 0; r < 4; ++r) {
            float v = fminf(fmaxf(bx[tid][r] / 1280.f, 0.f), 1.f);
            outp[tid * 4 + r] = kp ? v : 0.f;
        }
        outp[400 + tid] = kp ? tsc[tid] : 0.f;
        outp[500 + tid] = kp ? (float)lab[tid] : -1.f;
        outp[600 + tid] = (float)kp;
    }
}

// ---------------------------------------------------------------- launcher
extern "C" void kernel_launch(void* const* d_in, const int* in_sizes, int n_in,
                              void* d_out, int out_size, void* d_ws, size_t ws_size,
                              hipStream_t stream)
{
    (void)in_sizes; (void)n_in; (void)out_size; (void)ws_size;
    const float* c4  = (const float*)d_in[0];
    const float* c5  = (const float*)d_in[1];
    const float* w1a = (const float*)d_in[2];
    const float* s1a = (const float*)d_in[3];
    const float* b1a = (const float*)d_in[4];
    const float* w1b = (const float*)d_in[5];
    const float* s1b = (const float*)d_in[6];
    const float* b1b = (const float*)d_in[7];
    const float* wr  = (const float*)d_in[8];
    const float* sr  = (const float*)d_in[9];
    const float* br  = (const float*)d_in[10];
    const float* w2  = (const float*)d_in[11];
    const float* s2  = (const float*)d_in[12];
    const float* b2  = (const float*)d_in[13];
    const float* wp  = (const float*)d_in[14];
    const float* bp  = (const float*)d_in[15];

    char* ws = (char*)d_ws;
    bf16_t* wB1a   = (bf16_t*)(ws + OFF_WB1A);
    bf16_t* wB1b   = (bf16_t*)(ws + OFF_WB1B);
    bf16_t* wB2    = (bf16_t*)(ws + OFF_WB2);
    bf16_t* act0   = (bf16_t*)(ws + OFF_ACT0);
    bf16_t* act1   = (bf16_t*)(ws + OFF_ACT1);
    bf16_t* act3   = (bf16_t*)(ws + OFF_ACT3);
    bf16_t* act4   = (bf16_t*)(ws + OFF_ACT4);
    float*  predb  = (float*)(ws + OFF_PRED);
    float*  scrs   = (float*)(ws + OFF_SCORES);
    bf16_t* zrow   = (bf16_t*)(ws + OFF_ZROW);

    // zero halo row + weight prep (BN-scale folded, bf16, tap-major K)
    zero_bf16<<<(1280 + 255) / 256, 256, 0, stream>>>(zrow, 1280);
    {
        int t = 1024 * 1024 * 9;
        prep_weights<<<(t + 255) / 256, 256, 0, stream>>>(w1a, s1a, wB1a, 1024, t);
        prep_weights<<<(t + 255) / 256, 256, 0, stream>>>(w1b, s1b, wB1b, 1024, t);
        int t2 = 1024 * 1280 * 9;
        prep_weights<<<(t2 + 255) / 256, 256, 0, stream>>>(w2, s2, wB2, 1280, t2);
    }
    // c5 -> channels-last bf16
    {
        int t = 1024 * HW40;
        nchw_to_hwc_bf16<<<(t + 255) / 256, 256, 0, stream>>>(c5, act0, 1024, HW40, t);
    }
    // reorg path -> concat cols [0,256)
    reorg_concat<<<(1600 * 256) / 256, 256, 0, stream>>>(c4, wr, sr, br, act3);

    // three WMMA convs
    dim3 cg(25, 8);
    conv3x3_wmma<1024><<<cg, 256, 0, stream>>>(act0, wB1a, b1a, zrow, act1, 1024, 0);
    conv3x3_wmma<1024><<<cg, 256, 0, stream>>>(act1, wB1b, b1b, zrow, act3, 1280, 256);
    conv3x3_wmma<1280><<<cg, 256, 0, stream>>>(act3, wB2,  b2,  zrow, act4, 1024, 0);

    // head + postprocess
    {
        int t = HW40 * NPRED;
        pred_conv<<<(t + 255) / 256, 256, 0, stream>>>(act4, wp, bp, predb);
    }
    score_kernel<<<(8000 + 255) / 256, 256, 0, stream>>>(predb, scrs);
    post_kernel<<<1, 1024, 0, stream>>>(predb, scrs, (float*)d_out);
}
